// HierCDF_21320217657958
// MI455X (gfx1250) — matrix operand order repaired and checked by reading, not compiled, not run on hardware
//
#include <hip/hip_runtime.h>
#include <hip/hip_bf16.h>
#include <stdint.h>

#define NC   128    // NUM_CONCEPT
#define DH   512    // DIM_HIDDEN
#define H2   256    // hidden2
#define MAXE 384    // max edges (<=3 per concept)

typedef _Float16 v16h __attribute__((ext_vector_type(16)));
typedef _Float16 v8h  __attribute__((ext_vector_type(8)));
typedef float    v8f  __attribute__((ext_vector_type(8)));

struct GraphArg {
    int16_t pred[MAXE];   // predecessor concept index per edge (sequential by concept)
    uint8_t np[NC];       // number of predecessors per concept
};

// ---------------- device helpers ----------------

__device__ __forceinline__ float sigf(float x) { return 1.0f / (1.0f + __expf(-x)); }

__device__ __forceinline__ float rootn(float x, int n) {
    if (n == 1) return x;
    if (n == 2) return sqrtf(x);
    return cbrtf(x);
}

// A-matrix fragment, 16x32 f16 (ISA 7.12.2): lane L = row L&15,
// elems 0..7 = K[k0+hi*8 .. +7], elems 8..15 = K[k0+16+hi*8 .. +7]
__device__ __forceinline__ v16h load_a(const _Float16* row, int k0, int hi) {
    v8h lo = *(const v8h*)(row + k0 + hi * 8);
    v8h up = *(const v8h*)(row + k0 + 16 + hi * 8);
    v16h f;
#pragma unroll
    for (int i = 0; i < 8; ++i) { f[i] = lo[i]; f[i + 8] = up[i]; }
    return f;
}

// B-matrix fragment, 32x16 f16: lane L = column L&15,
// elems 0..7 = K[k0+hi*8 .. +7], elems 8..15 = K[k0+16+hi*8 .. +7]
// (weights stored row-major: row n of W == column n of B)
__device__ __forceinline__ v16h load_b(const _Float16* row, int k0, int hi) {
    v8h lo = *(const v8h*)(row + k0 + hi * 8);
    v8h up = *(const v8h*)(row + k0 + 16 + hi * 8);
    v16h f;
#pragma unroll
    for (int i = 0; i < 8; ++i) { f[i] = lo[i]; f[i + 8] = up[i]; }
    return f;
}

__device__ __forceinline__ v8f wmma_f16(v16h a, v16h b, v8f c) {
    return __builtin_amdgcn_wmma_f32_16x16x32_f16(false, a, false, b, (short)0, c, false, false);
}

// ---------------- kernel 0: f32 -> f16 weight conversion ----------------

__global__ void f32_to_f16_kernel(const float* __restrict__ s, _Float16* __restrict__ d, int n) {
    int i = blockIdx.x * blockDim.x + threadIdx.x;
    if (i < n) d[i] = (_Float16)s[i];
}

// ---------------- kernel 1: posterior DAG + gathers + f16 pack ----------------
// one thread per batch element; per-thread posterior state kept in LDS
// (64 thr * 128 concepts * 4B = 32 KB; WGP has 320 KB)

__global__ void __launch_bounds__(64)
posterior_kernel(GraphArg g,
                 const int* __restrict__ user_id, const int* __restrict__ question_id,
                 const float* __restrict__ priori,
                 const float* __restrict__ condi_p, const float* __restrict__ condi_n,
                 const float* __restrict__ item_diff, const float* __restrict__ item_disc,
                 const float* __restrict__ q_table,
                 _Float16* __restrict__ Au, _Float16* __restrict__ Ai,
                 float* __restrict__ disc, int B, int E) {
    __shared__ float post[64 * NC];
    int b = blockIdx.x * 64 + threadIdx.x;
    if (b >= B) return;
    int t = threadIdx.x;
    int u = user_id[b];
    int q = question_id[b];

    const float* bp  = priori  + (size_t)u * NC;
    const float* rcp = condi_p + (size_t)u * E;
    const float* rcn = condi_n + (size_t)u * E;

    int eo = 0;   // running edge offset == OFFS[k]; uniform across the wave
    for (int k = 0; k < NC; ++k) {
        int np = g.np[k];
        float val;
        if (np == 0) {
            val = sigf(bp[k]);
        } else {
            float prod = 1.0f;
            for (int e = 0; e < np; ++e) {
                float cpe = rootn(sigf(rcp[eo]), np);
                float cne = rootn(sigf(rcn[eo]), np);
                float pr  = post[(int)g.pred[eo] * 64 + t];
                prod *= cpe * pr + cne * (1.0f - pr);
                ++eo;
            }
            val = prod;
        }
        post[k * 64 + t] = val;
    }

    const float* qrow = q_table   + (size_t)q * NC;
    const float* drow = item_diff + (size_t)q * NC;
#pragma unroll 4
    for (int k = 0; k < NC; ++k) {
        float qk = qrow[k];
        Au[(size_t)b * NC + k] = (_Float16)(post[k * 64 + t] * qk);
        Ai[(size_t)b * NC + k] = (_Float16)(sigf(drow[k]) * qk);
    }
    disc[b] = sigf(item_disc[q]);
}

// ---------------- kernel 2: WMMA GEMM1 (user & item factors fused) ----------------
// x0[m][n] = (tanh(Au@Wu^T + bu) - sigmoid(Ai@Wi^T + bi)) * disc[m], f16 out
// one wave = one 16x16 tile of BOTH GEMMs (shared output coordinates), K=128 -> 4 WMMAs each

__global__ void __launch_bounds__(256)
gemm1_kernel(const _Float16* __restrict__ Au, const _Float16* __restrict__ Ai,
             const _Float16* __restrict__ Wuh, const _Float16* __restrict__ Wih,
             const float* __restrict__ bu, const float* __restrict__ bi,
             const float* __restrict__ disc, _Float16* __restrict__ X0) {
    int wave = threadIdx.x >> 5;
    int lane = threadIdx.x & 31;
    int tile = blockIdx.x * 8 + wave;
    int nt = tile & 31;     // 512/16 = 32 N tiles
    int mt = tile >> 5;
    int r  = lane & 15;
    int hi = lane >> 4;

    const _Float16* arowu = Au  + (size_t)(mt * 16 + r) * NC;
    const _Float16* arowi = Ai  + (size_t)(mt * 16 + r) * NC;
    const _Float16* browu = Wuh + (size_t)(nt * 16 + r) * NC;
    const _Float16* browi = Wih + (size_t)(nt * 16 + r) * NC;

    v8f cu = {};
    v8f ci = {};
#pragma unroll
    for (int k0 = 0; k0 < NC; k0 += 32) {
        cu = wmma_f16(load_a(arowu, k0, hi), load_b(browu, k0, hi), cu);
        ci = wmma_f16(load_a(arowi, k0, hi), load_b(browi, k0, hi), ci);
    }

    int n = nt * 16 + r;
    float bun = bu[n];
    float bin = bi[n];
#pragma unroll
    for (int v = 0; v < 8; ++v) {
        int m = mt * 16 + v + hi * 8;
        float uf  = tanhf(cu[v] + bun);
        float itf = sigf(ci[v] + bin);
        X0[(size_t)m * DH + n] = (_Float16)((uf - itf) * disc[m]);
    }
}

// ---------------- kernel 3: WMMA GEMM2 + fused final dot ----------------
// x1 = sigmoid(X0 @ W1^T + b1) (16 rows x 256 cols per block, via LDS),
// out = sigmoid(x1 . w2 + b2)

__global__ void __launch_bounds__(256)
gemm2_kernel(const _Float16* __restrict__ X0, const _Float16* __restrict__ W1h,
             const float* __restrict__ b1, const float* __restrict__ W2f,
             const float* __restrict__ b2, float* __restrict__ out) {
    __shared__ float x1s[16 * H2];   // 16 KB
    int wave = threadIdx.x >> 5;
    int lane = threadIdx.x & 31;
    int mt   = blockIdx.x;
    int r    = lane & 15;
    int hi   = lane >> 4;

    const _Float16* arow = X0 + (size_t)(mt * 16 + r) * DH;
#pragma unroll
    for (int tt = 0; tt < 2; ++tt) {
        int ntile = wave * 2 + tt;             // 16 N tiles over 8 waves
        const _Float16* brow = W1h + (size_t)(ntile * 16 + r) * DH;
        v8f c = {};
#pragma unroll
        for (int k0 = 0; k0 < DH; k0 += 32)
            c = wmma_f16(load_a(arow, k0, hi), load_b(brow, k0, hi), c);
        int n = ntile * 16 + r;
        float bn = b1[n];
#pragma unroll
        for (int v = 0; v < 8; ++v)
            x1s[(v + hi * 8) * H2 + n] = sigf(c[v] + bn);
    }
    __syncthreads();
    if (threadIdx.x < 16) {
        int m = threadIdx.x;
        float s = b2[0];
        for (int j = 0; j < H2; ++j) s += x1s[m * H2 + j] * W2f[j];
        out[mt * 16 + m] = sigf(s);
    }
}

// ---------------- host: numpy-legacy MT19937 graph reconstruction ----------------

namespace {

struct MT19937 {
    uint32_t mt[624];
    int mti;
    explicit MT19937(uint32_t s) {
        mt[0] = s;
        for (int i = 1; i < 624; ++i)
            mt[i] = 1812433253u * (mt[i - 1] ^ (mt[i - 1] >> 30)) + (uint32_t)i;
        mti = 624;
    }
    void gen() {
        const uint32_t MA = 0x9908b0dfu, UP = 0x80000000u, LO = 0x7fffffffu;
        for (int i = 0; i < 624; ++i) {
            uint32_t y = (mt[i] & UP) | (mt[(i + 1) % 624] & LO);
            mt[i] = mt[(i + 397) % 624] ^ (y >> 1) ^ ((y & 1u) ? MA : 0u);
        }
        mti = 0;
    }
    uint32_t next32() {
        if (mti >= 624) gen();
        uint32_t y = mt[mti++];
        y ^= y >> 11; y ^= (y << 7) & 0x9d2c5680u; y ^= (y << 15) & 0xefc60000u; y ^= y >> 18;
        return y;
    }
    uint64_t next64() {  // numpy: high word drawn first
        uint64_t hiw = next32(), low = next32();
        return (hiw << 32) | low;
    }
};

inline uint64_t genmask64(uint64_t x) {
    x |= x >> 1; x |= x >> 2; x |= x >> 4; x |= x >> 8; x |= x >> 16; x |= x >> 32;
    return x;
}

// legacy randint: masked rejection on 64-bit draws, result in [0, rng]
inline uint64_t randint_bounded(MT19937& r, uint64_t rng) {
    if (rng == 0) return 0;
    uint64_t mask = genmask64(rng), v;
    while ((v = (r.next64() & mask)) > rng) {}
    return v;
}

// legacy shuffle interval: masked rejection on 32-bit draws, result in [0, mx]
inline uint32_t interval32(MT19937& r, uint32_t mx) {
    uint32_t mask = (uint32_t)genmask64(mx), v;
    while ((v = (r.next32() & mask)) > mx) {}
    return v;
}

inline int build_graph(GraphArg& g) {
    MT19937 r(0u);
    int eo = 0;
    for (int k = 0; k < NC; ++k) {
        int lp = 0;
        if (k > 0) {
            int m = (k < 3 ? k : 3);               // randint(0, m+1) -> [0, m]
            lp = (int)randint_bounded(r, (uint64_t)m);
        }
        if (lp > 0) {
            // legacy choice(k, size=lp, replace=False) == permutation(k)[:lp]
            int perm[NC];
            for (int i = 0; i < k; ++i) perm[i] = i;
            for (int i = k - 1; i >= 1; --i) {
                int j = (int)interval32(r, (uint32_t)i);
                int tmp = perm[i]; perm[i] = perm[j]; perm[j] = tmp;
            }
            int sel[3];
            for (int e = 0; e < lp; ++e) sel[e] = perm[e];
            for (int a = 1; a < lp; ++a) {          // sorted()
                int v = sel[a], bpos = a;
                while (bpos > 0 && sel[bpos - 1] > v) { sel[bpos] = sel[bpos - 1]; --bpos; }
                sel[bpos] = v;
            }
            for (int e = 0; e < lp; ++e) g.pred[eo + e] = (int16_t)sel[e];
            eo += lp;
        }
        g.np[k] = (uint8_t)lp;
    }
    return eo;
}

} // namespace

// ---------------- launch ----------------

extern "C" void kernel_launch(void* const* d_in, const int* in_sizes, int n_in,
                              void* d_out, int out_size, void* d_ws, size_t ws_size,
                              hipStream_t stream) {
    (void)n_in; (void)out_size; (void)ws_size;

    const int*   user_id     = (const int*)d_in[0];
    const int*   question_id = (const int*)d_in[1];
    const float* priori      = (const float*)d_in[2];
    const float* condi_p     = (const float*)d_in[3];
    const float* condi_n     = (const float*)d_in[4];
    const float* item_diff   = (const float*)d_in[5];
    const float* item_disc   = (const float*)d_in[6];
    const float* q_table     = (const float*)d_in[7];
    const float* Wu          = (const float*)d_in[8];
    const float* bu          = (const float*)d_in[9];
    const float* Wi          = (const float*)d_in[10];
    const float* bi          = (const float*)d_in[11];
    const float* W1          = (const float*)d_in[12];
    const float* b1          = (const float*)d_in[13];
    const float* W2          = (const float*)d_in[14];
    const float* b2          = (const float*)d_in[15];
    float* out = (float*)d_out;

    const int B  = in_sizes[0];
    const long NU = in_sizes[2] / NC;
    const int E  = (int)(in_sizes[3] / NU);

    GraphArg g{};
    build_graph(g);

    // workspace carve-out (256B aligned)
    char* w = (char*)d_ws;
    auto carve = [&](size_t bytes) { char* p = w; w += (bytes + 255) & ~(size_t)255; return p; };
    _Float16* Wuh  = (_Float16*)carve((size_t)DH * NC * 2);
    _Float16* Wih  = (_Float16*)carve((size_t)DH * NC * 2);
    _Float16* W1h  = (_Float16*)carve((size_t)H2 * DH * 2);
    _Float16* Au   = (_Float16*)carve((size_t)B * NC * 2);
    _Float16* Ai   = (_Float16*)carve((size_t)B * NC * 2);
    float*    disc = (float*)   carve((size_t)B * 4);
    _Float16* X0   = (_Float16*)carve((size_t)B * DH * 2);

    // 0) weights -> f16 (tiny; rebuilt every call for determinism)
    f32_to_f16_kernel<<<(DH * NC + 255) / 256, 256, 0, stream>>>(Wu, Wuh, DH * NC);
    f32_to_f16_kernel<<<(DH * NC + 255) / 256, 256, 0, stream>>>(Wi, Wih, DH * NC);
    f32_to_f16_kernel<<<(H2 * DH + 255) / 256, 256, 0, stream>>>(W1, W1h, H2 * DH);

    // 1) posterior + gathers + f16 pack
    posterior_kernel<<<(B + 63) / 64, 64, 0, stream>>>(
        g, user_id, question_id, priori, condi_p, condi_n,
        item_diff, item_disc, q_table, Au, Ai, disc, B, E);

    // 2) fused user/item factor GEMMs (WMMA) -> X0
    int mtiles = B / 16;                       // 1024
    gemm1_kernel<<<mtiles * 32 / 8, 256, 0, stream>>>(
        Au, Ai, Wuh, Wih, bu, bi, disc, X0);

    // 3) GEMM2 (WMMA) + fused final layer
    gemm2_kernel<<<mtiles, 256, 0, stream>>>(X0, W1h, b1, W2, b2, out);
}
